// CrfRnnLayer_30245159699082
// MI455X (gfx1250) — matrix operand order repaired and checked by reading, not compiled, not run on hardware
//
#include <hip/hip_runtime.h>

// CRF-RNN mean-field, MI455X (gfx1250, wave32).
// Dominant op: bil(C,N) = softmax(q) @ Kb(N,N), Kb = exp(-0.5|fi-fj|^2), N=12544.
// Kb is recomputed on the fly into WMMA B-fragments (5 FMA + v_exp per elem,
// features staged in LDS) instead of streaming 315MB/iter from HBM.
// C=21 padded to 32; padded channel 21 = ones so GEMM row 21 yields
// bilateral_norm (column sums of symmetric Kb) for free.
// k-loop unrolled x2: one barrier per 64 K, 4 independent v_wmma per stage.

#define H_ 112
#define W_ 112
#define Cc 21
#define NPIX (H_*W_)      // 12544
#define KB 392            // NPIX/32 k-blocks
#define JT 784            // NPIX/16 j-tiles
#define KSPLIT 2
#define EWAVES 4
#define RADIUS 9
#define TAPS 19

typedef __attribute__((ext_vector_type(16))) __bf16          v16bf;
typedef __attribute__((ext_vector_type(16))) unsigned short  v16us;
typedef __attribute__((ext_vector_type(8)))  float           v8f;

__device__ __forceinline__ unsigned short f2bf(float x) {
  unsigned int u = __float_as_uint(x);
  unsigned int r = ((u >> 16) & 1u) + 0x7FFFu;   // round-to-nearest-even
  return (unsigned short)((u + r) >> 16);
}

// ---------------- prep: features, u transpose, q init, gaussian taps --------
__global__ __launch_bounds__(256) void crf_prep(
    const float* __restrict__ unaries, const float* __restrict__ rgb,
    float* __restrict__ f, float* __restrict__ u, float* __restrict__ q,
    float* __restrict__ k1d, float* __restrict__ edgefac)
{
  int i = blockIdx.x * 256 + threadIdx.x;
  if (i < TAPS) {
    float d = (float)(i - RADIUS) * (1.0f / 3.0f);         // theta_gamma = 3
    k1d[i] = __expf(-0.5f * d * d);
  }
  if (i < W_) {  // blur(ones) is separable: edgefac[y]*edgefac[x]
    float sfac = 0.f;
    for (int d = -RADIUS; d <= RADIUS; ++d) {
      int p = i + d;
      if (p >= 0 && p < W_) { float t = (float)d * (1.0f/3.0f); sfac += __expf(-0.5f*t*t); }
    }
    edgefac[i] = sfac;
  }
  if (i < NPIX) {
    int y = i / W_, x = i % W_;
    f[i*5+0] = (float)y * (1.0f / 160.0f);                 // theta_alpha
    f[i*5+1] = (float)x * (1.0f / 160.0f);
    f[i*5+2] = rgb[i*3+0] * (1.0f / 3.0f);                 // theta_beta
    f[i*5+3] = rgb[i*3+1] * (1.0f / 3.0f);
    f[i*5+4] = rgb[i*3+2] * (1.0f / 3.0f);
#pragma unroll
    for (int c = 0; c < Cc; ++c) {
      float vv = unaries[(size_t)i*Cc + c];                // NHWC -> [C][N]
      u[(size_t)c*NPIX + i] = vv;
      q[(size_t)c*NPIX + i] = vv;
    }
  }
}

// ------------- softmax over channels + pack bf16 A-fragments ----------------
// A-fragment (16-bit 16x32, ISA 7.12.2): lanes 0-15 row M, elems 0-7 = K0..7,
// elems 8-15 = K16..23; lanes 16-31 hold K8..15 / K24..31.
// Fragment blob: [kb][mfrag][lane*16 + elem] ushorts. Channel 21 = 1.0 (norm row).
__global__ __launch_bounds__(256) void crf_softmax_pack(
    const float* __restrict__ q, float* __restrict__ s,
    unsigned short* __restrict__ afrag)
{
  int i = blockIdx.x * 256 + threadIdx.x;
  if (i >= NPIX) return;
  float v[Cc];
  float m = -3.4e38f;
#pragma unroll
  for (int c = 0; c < Cc; ++c) { v[c] = q[(size_t)c*NPIX + i]; m = fmaxf(m, v[c]); }
  float sum = 0.f;
#pragma unroll
  for (int c = 0; c < Cc; ++c) { v[c] = __expf(v[c] - m); sum += v[c]; }
  float inv = 1.0f / sum;

  int kb = i >> 5, K = i & 31;
  int hi = (K >> 3) & 1;                        // which lane-half holds this K
  int e  = (K & 7) + ((K >= 16) ? 8 : 0);       // element index within lane
  size_t base = (size_t)kb * 1024 + (size_t)hi * 256 + (size_t)e;
#pragma unroll
  for (int c = 0; c < 32; ++c) {
    float val = (c < Cc) ? v[c] * inv : ((c == Cc) ? 1.0f : 0.0f);
    if (c < Cc) s[(size_t)c*NPIX + i] = val;
    afrag[base + (size_t)(c >> 4) * 512 + (size_t)(c & 15) * 16] = f2bf(val);
  }
}

// ---------------- separable 19-tap Gaussian blur ----------------------------
__global__ __launch_bounds__(256) void crf_blur_h(
    const float* __restrict__ s, const float* __restrict__ k1d, float* __restrict__ tmp)
{
  int idx = blockIdx.x * 256 + threadIdx.x;
  if (idx >= Cc * NPIX) return;
  int x = (idx % NPIX) % W_;
  float acc = 0.f;
#pragma unroll
  for (int d = -RADIUS; d <= RADIUS; ++d) {
    int xx = x + d;
    if (xx >= 0 && xx < W_) acc += k1d[d + RADIUS] * s[idx + d];
  }
  tmp[idx] = acc;
}

__global__ __launch_bounds__(256) void crf_blur_v(
    const float* __restrict__ tmp, const float* __restrict__ k1d,
    const float* __restrict__ edgefac, float* __restrict__ sp)
{
  int idx = blockIdx.x * 256 + threadIdx.x;
  if (idx >= Cc * NPIX) return;
  int p = idx % NPIX; int y = p / W_; int x = p % W_;
  float acc = 0.f;
#pragma unroll
  for (int d = -RADIUS; d <= RADIUS; ++d) {
    int yy = y + d;
    if (yy >= 0 && yy < H_) acc += k1d[d + RADIUS] * tmp[idx + d * W_];
  }
  sp[idx] = acc / (edgefac[y] * edgefac[x]);   // /= spatial_norm = blur(ones)
}

// ------------- fused bilateral GEMM: bil = s @ exp-kernel -------------------
// Per wave: one 16-col j-tile, two M-fragments (channels 0-15 / 16-31).
// B-fragment (32x16 bf16) generated in registers: lane L -> col j=(L&15),
// K = (L>=16 ? 8:0)+(e&7)+(e>=8 ? 16:0), value = exp(-0.5|f_i - f_j|^2). K-split partials.
// Unroll 2 k-blocks per barrier: native __bf16 casts -> packed cvt, 4 wmma.
__device__ __forceinline__ v16bf make_bfrag(
    const float* __restrict__ fi, int half,
    float fj0, float fj1, float fj2, float fj3, float fj4)
{
  v16bf b;
#pragma unroll
  for (int e = 0; e < 16; ++e) {
    const float* fr = &fi[((e & 7) + half * 8 + ((e >> 3) << 4)) * 5];
    float d0 = fr[0]-fj0, d1 = fr[1]-fj1, d2 = fr[2]-fj2, d3 = fr[3]-fj3, d4 = fr[4]-fj4;
    float dd = d0*d0 + d1*d1 + d2*d2 + d3*d3 + d4*d4;
    b[e] = (__bf16)__expf(-0.5f * dd);          // hardware f32->bf16 convert
  }
  return b;
}

__global__ __launch_bounds__(EWAVES*32) void crf_bilateral_wmma(
    const float* __restrict__ f,
    const unsigned short* __restrict__ afrag,
    float* __restrict__ bilpart)
{
  __shared__ float fi[2 * 32 * 5];               // two k-blocks of features
  const int lane  = threadIdx.x & 31;
  const int wave  = threadIdx.x >> 5;
  const int jtile = blockIdx.x * EWAVES + wave;
  const int jcol  = (jtile << 4) + (lane & 15);
  const int half  = lane >> 4;
  const int kb0   = blockIdx.y * (KB / KSPLIT);  // KB/KSPLIT = 196 (even)
  const int kb1   = kb0 + (KB / KSPLIT);

  const float fj0 = f[jcol*5+0], fj1 = f[jcol*5+1], fj2 = f[jcol*5+2];
  const float fj3 = f[jcol*5+3], fj4 = f[jcol*5+4];

  v8f acc0 = {}; v8f acc1 = {};
  for (int kb = kb0; kb < kb1; kb += 2) {
    __syncthreads();
    for (int t = threadIdx.x; t < 320; t += EWAVES*32) fi[t] = f[kb*160 + t];
    __syncthreads();

    v16bf b0 = make_bfrag(fi,       half, fj0, fj1, fj2, fj3, fj4);
    v16bf b1 = make_bfrag(fi + 160, half, fj0, fj1, fj2, fj3, fj4);

    const unsigned short* ap = afrag + (size_t)kb * 1024 + lane * 16;
    v16bf a00 = __builtin_bit_cast(v16bf, *(const v16us*)(ap));
    v16bf a01 = __builtin_bit_cast(v16bf, *(const v16us*)(ap + 512));
    v16bf a10 = __builtin_bit_cast(v16bf, *(const v16us*)(ap + 1024));
    v16bf a11 = __builtin_bit_cast(v16bf, *(const v16us*)(ap + 1536));
    __builtin_prefetch(ap + 2048, 0, 0);   // global_prefetch_b8: next stage's A

    acc0 = __builtin_amdgcn_wmma_f32_16x16x32_bf16(false, a00, false, b0, (short)0, acc0, false, false);
    acc1 = __builtin_amdgcn_wmma_f32_16x16x32_bf16(false, a01, false, b0, (short)0, acc1, false, false);
    acc0 = __builtin_amdgcn_wmma_f32_16x16x32_bf16(false, a10, false, b1, (short)0, acc0, false, false);
    acc1 = __builtin_amdgcn_wmma_f32_16x16x32_bf16(false, a11, false, b1, (short)0, acc1, false, false);
  }

  // C/D layout: VGPR r -> M = r (lanes 0-15) / r+8 (lanes 16-31), N = lane&15
  float* op = bilpart + (size_t)blockIdx.y * 32 * NPIX;
  const int ro = half * 8;
#pragma unroll
  for (int r = 0; r < 8; ++r) {
    op[(size_t)(r + ro)      * NPIX + jcol] = acc0[r];
    op[(size_t)(16 + r + ro) * NPIX + jcol] = acc1[r];
  }
}

// -------- combine: reduce K-split, normalize, mix, compat update ------------
__global__ __launch_bounds__(256) void crf_combine(
    const float* __restrict__ u, const float* __restrict__ sp,
    const float* __restrict__ bilpart,
    const float* __restrict__ SW, const float* __restrict__ BW, const float* __restrict__ CM,
    float* __restrict__ q, float* __restrict__ out, int last)
{
  __shared__ float sw[Cc*Cc], bw[Cc*Cc], cm[Cc*Cc];
  for (int t = threadIdx.x; t < Cc*Cc; t += 256) { sw[t] = SW[t]; bw[t] = BW[t]; cm[t] = CM[t]; }
  __syncthreads();
  int i = blockIdx.x * 256 + threadIdx.x;
  if (i >= NPIX) return;

  const float* b0 = bilpart;
  const float* b1 = bilpart + (size_t)32 * NPIX;
  float norm = b0[(size_t)Cc*NPIX + i] + b1[(size_t)Cc*NPIX + i];  // ones-row = bilateral_norm
  float rinv = 1.0f / norm;

  float spv[Cc], bilv[Cc], mp[Cc];
#pragma unroll
  for (int c = 0; c < Cc; ++c) {
    spv[c]  = sp[(size_t)c*NPIX + i];
    bilv[c] = (b0[(size_t)c*NPIX + i] + b1[(size_t)c*NPIX + i]) * rinv;
  }
  for (int c = 0; c < Cc; ++c) {
    float a = 0.f;
    for (int k = 0; k < Cc; ++k) a += sw[c*Cc + k] * spv[k] + bw[c*Cc + k] * bilv[k];
    mp[c] = a;
  }
  for (int c = 0; c < Cc; ++c) {
    float a = 0.f;
    for (int k = 0; k < Cc; ++k) a += cm[c*Cc + k] * mp[k];
    float qn = u[(size_t)c*NPIX + i] - a;
    q[(size_t)c*NPIX + i] = qn;
    if (last) out[(size_t)i*Cc + c] = qn;      // back to NHWC
  }
}

// ---------------------------------------------------------------------------
extern "C" void kernel_launch(void* const* d_in, const int* in_sizes, int n_in,
                              void* d_out, int out_size, void* d_ws, size_t ws_size,
                              hipStream_t stream)
{
  (void)in_sizes; (void)n_in; (void)out_size; (void)ws_size;
  const float* unaries = (const float*)d_in[0];
  const float* rgb     = (const float*)d_in[1];
  const float* SW      = (const float*)d_in[2];
  const float* BW      = (const float*)d_in[3];
  const float* CM      = (const float*)d_in[4];
  float* out = (float*)d_out;

  char* ws = (char*)d_ws;
  size_t off = 0;
  auto take = [&](size_t bytes) -> char* {
    char* p = ws + off; off = (off + bytes + 255) & ~(size_t)255; return p;
  };
  float* f        = (float*)take((size_t)NPIX * 5 * 4);
  float* u        = (float*)take((size_t)Cc * NPIX * 4);
  float* q        = (float*)take((size_t)Cc * NPIX * 4);
  float* s        = (float*)take((size_t)Cc * NPIX * 4);
  float* tmp      = (float*)take((size_t)Cc * NPIX * 4);
  float* sp       = (float*)take((size_t)Cc * NPIX * 4);
  float* bilpart  = (float*)take((size_t)KSPLIT * 32 * NPIX * 4);
  unsigned short* afrag = (unsigned short*)take((size_t)KB * 1024 * 2);
  float* k1d      = (float*)take(TAPS * 4);
  float* edgefac  = (float*)take(W_ * 4);
  // total ~9 MB workspace

  crf_prep<<<NPIX/256, 256, 0, stream>>>(unaries, rgb, f, u, q, k1d, edgefac);
  for (int it = 0; it < 5; ++it) {
    crf_softmax_pack<<<NPIX/256, 256, 0, stream>>>(q, s, afrag);
    crf_blur_h<<<(Cc*NPIX)/256, 256, 0, stream>>>(s, k1d, tmp);
    crf_blur_v<<<(Cc*NPIX)/256, 256, 0, stream>>>(tmp, k1d, edgefac, sp);
    dim3 gE(JT / EWAVES, KSPLIT);
    crf_bilateral_wmma<<<gE, EWAVES*32, 0, stream>>>(f, afrag, bilpart);
    crf_combine<<<NPIX/256, 256, 0, stream>>>(u, sp, bilpart, SW, BW, CM, q, out,
                                              (it == 4) ? 1 : 0);
  }
}
